// HPUMLAImpl_64484638982545
// MI455X (gfx1250) — compile-verified, matrix operand order checked by reading, not run on hardware
//
#include <hip/hip_runtime.h>
#include <hip/hip_bf16.h>

#define SCALE_F 0.07216878364870322f
#define BATCH 32
#define NUM_BLOCKS 1024
#define BLOCK_SZ 128
#define NH 16
#define KV_LORA 512
#define ROPE 64
#define DTOT 576
#define V_HEAD 128
#define D_MODEL 2048

// LDS row strides in halfs (multiples of 8 => 16B-aligned b128 reads)
#define QSTR 584
#define VSTR 136
#define PSTR 136
#define WTSTR 520
#define USTR 2056

typedef _Float16 h8   __attribute__((ext_vector_type(8)));
typedef _Float16 v16h __attribute__((ext_vector_type(16)));
typedef float    v8f  __attribute__((ext_vector_type(8)));

__device__ __forceinline__ v16h cat16(h8 lo, h8 hi) {
    v16h r;
#pragma unroll
    for (int i = 0; i < 8; ++i) { r[i] = lo[i]; r[i + 8] = hi[i]; }
    return r;
}
// A/B fragment from LDS: 8 contiguous halfs at p, 8 more at p+16
__device__ __forceinline__ v16h frag_lds(const _Float16* p) {
    return cat16(*(const h8*)p, *(const h8*)(p + 16));
}
// fragment from 4 float4s (k..k+7 and k+16..k+23), converted to f16
__device__ __forceinline__ v16h frag_f32(float4 a, float4 b, float4 c, float4 d) {
    v16h r;
    r[0]=(_Float16)a.x; r[1]=(_Float16)a.y; r[2]=(_Float16)a.z;  r[3]=(_Float16)a.w;
    r[4]=(_Float16)b.x; r[5]=(_Float16)b.y; r[6]=(_Float16)b.z;  r[7]=(_Float16)b.w;
    r[8]=(_Float16)c.x; r[9]=(_Float16)c.y; r[10]=(_Float16)c.z; r[11]=(_Float16)c.w;
    r[12]=(_Float16)d.x;r[13]=(_Float16)d.y;r[14]=(_Float16)d.z; r[15]=(_Float16)d.w;
    return r;
}
__device__ __forceinline__ v8f wmma_f16(v16h A, v16h B, v8f C) {
    return __builtin_amdgcn_wmma_f32_16x16x32_f16(false, A, false, B, (short)0, C,
                                                  false, false);
}

// ---------------------------------------------------------------------------
// Kernel 1: per-KV-block attention. 1 workgroup (256 thr / 8 waves) per block.
// ---------------------------------------------------------------------------
__global__ __launch_bounds__(256) void mla_block_attn(
    const float* __restrict__ q_nope, const float* __restrict__ q_pe,
    const float* __restrict__ key_cache, const float* __restrict__ value_cache,
    const float* __restrict__ block_bias, const int* __restrict__ block_list,
    const int* __restrict__ block_groups,
    float* __restrict__ o_part, float* __restrict__ bmax_out,
    float* __restrict__ bsum_out)
{
    __shared__ __align__(16) _Float16 qh[NH * QSTR];        // 18,688 B
    __shared__ __align__(16) _Float16 vT[KV_LORA * VSTR];   // 139,264 B (v transposed)
    __shared__ __align__(16) _Float16 pl[NH * PSTR];        // 4,352 B
    __shared__ float redmax[8 * NH];
    __shared__ float redsum[8 * NH];

    const int n    = blockIdx.x;
    const int tid  = threadIdx.x;
    const int wave = tid >> 5;
    const int lane = tid & 31;
    const int b    = block_groups[n];
    const int blk  = block_list[n];

    // ---- stage q (scaled) as f16 ----
    for (int idx = tid; idx < NH * DTOT; idx += 256) {
        int h = idx / DTOT, d = idx - h * DTOT;
        float v = (d < KV_LORA)
                      ? q_nope[((size_t)b * NH + h) * KV_LORA + d]
                      : q_pe[((size_t)b * NH + h) * ROPE + (d - KV_LORA)];
        qh[h * QSTR + d] = (_Float16)(v * SCALE_F);
    }
    // ---- stage v transposed as f16 (vT[d][s]) ----
    const float* vsrc = value_cache + (size_t)blk * BLOCK_SZ * KV_LORA;
    for (int e4 = tid; e4 < BLOCK_SZ * KV_LORA / 4; e4 += 256) {
        int e = e4 * 4, s = e >> 9, d = e & 511;
        float4 f = *(const float4*)(vsrc + e);
        vT[(d + 0) * VSTR + s] = (_Float16)f.x;
        vT[(d + 1) * VSTR + s] = (_Float16)f.y;
        vT[(d + 2) * VSTR + s] = (_Float16)f.z;
        vT[(d + 3) * VSTR + s] = (_Float16)f.w;
    }
    __syncthreads();

    const int crow = lane & 15;  // A row / tile column
    const int hi   = lane >> 4;
    const int kb   = hi * 8;
    const int scol = wave * 16 + crow;  // sequence position for QK

    // ---- QK: attn[h, scol], K = 576 in 18 chunks of 32 ----
    const float* vrow = value_cache + ((size_t)blk * BLOCK_SZ + scol) * KV_LORA;
    const float* krow = key_cache + ((size_t)blk * BLOCK_SZ + scol) * ROPE;
    v8f acc = {};
#pragma unroll
    for (int kc = 0; kc < DTOT / 32; ++kc) {
        v16h A = frag_lds(&qh[crow * QSTR + kc * 32 + kb]);
        const float* src;
        int doff;
        if (kc < 16) { src = vrow; doff = kc * 32 + kb; }
        else         { src = krow; doff = (kc - 16) * 32 + kb; }
        float4 f0 = *(const float4*)(src + doff);
        float4 f1 = *(const float4*)(src + doff + 4);
        float4 f2 = *(const float4*)(src + doff + 16);
        float4 f3 = *(const float4*)(src + doff + 20);
        acc = wmma_f16(A, frag_f32(f0, f1, f2, f3), acc);
    }
    float bias = block_bias[(size_t)n * BLOCK_SZ + scol];
    float sc[8];
#pragma unroll
    for (int j = 0; j < 8; ++j) sc[j] = acc[j] + bias;

    // ---- per-head max over 128 positions ----
    float mx[8];
#pragma unroll
    for (int j = 0; j < 8; ++j) {
        float m = sc[j];
        for (int d = 8; d >= 1; d >>= 1) m = fmaxf(m, __shfl_xor(m, d, 32));
        mx[j] = m;
    }
    if (crow == 0)
#pragma unroll
        for (int j = 0; j < 8; ++j) redmax[wave * 16 + hi * 8 + j] = mx[j];
    __syncthreads();
    float hm[8];
#pragma unroll
    for (int j = 0; j < 8; ++j) {
        float m = -3.402823466e38f;
        for (int w = 0; w < 8; ++w) m = fmaxf(m, redmax[w * 16 + hi * 8 + j]);
        hm[j] = m;
    }
    // ---- p = exp(attn - max); stash f16 p, reduce sums ----
    float sm[8];
#pragma unroll
    for (int j = 0; j < 8; ++j) {
        float p = __expf(sc[j] - hm[j]);
        sm[j] = p;
        pl[(hi * 8 + j) * PSTR + scol] = (_Float16)p;
    }
#pragma unroll
    for (int j = 0; j < 8; ++j) {
        float s = sm[j];
        for (int d = 8; d >= 1; d >>= 1) s += __shfl_xor(s, d, 32);
        sm[j] = s;
    }
    if (crow == 0)
#pragma unroll
        for (int j = 0; j < 8; ++j) redsum[wave * 16 + hi * 8 + j] = sm[j];
    __syncthreads();
    if (tid < NH) {
        float s = 0.f, m = -3.402823466e38f;
        for (int w = 0; w < 8; ++w) {
            s += redsum[w * 16 + tid];
            m = fmaxf(m, redmax[w * 16 + tid]);
        }
        bsum_out[(size_t)n * NH + tid] = s;
        bmax_out[(size_t)n * NH + tid] = m;
    }

    // ---- PV: o(16x512) = p(16x128) x v(128x512); 4 N-tiles per wave ----
#pragma unroll
    for (int t = 0; t < 4; ++t) {
        int col = (wave * 4 + t) * 16 + crow;  // v index
        v8f acc2 = {};
#pragma unroll
        for (int kc = 0; kc < 4; ++kc) {
            v16h A = frag_lds(&pl[crow * PSTR + kc * 32 + kb]);
            v16h B = frag_lds(&vT[col * VSTR + kc * 32 + kb]);
            acc2 = wmma_f16(A, B, acc2);
        }
#pragma unroll
        for (int j = 0; j < 8; ++j) {
            int h = hi * 8 + j;
            o_part[((size_t)n * NH + h) * KV_LORA + col] = acc2[j];
        }
    }
}

// ---------------------------------------------------------------------------
// Kernel 2: LSE combine across the 32 blocks of each (batch, head)
// ---------------------------------------------------------------------------
__global__ __launch_bounds__(128) void mla_combine(
    const float* __restrict__ o_part, const float* __restrict__ bmax,
    const float* __restrict__ bsum, float* __restrict__ out_comb)
{
    const int g = blockIdx.x;
    const int b = g >> 4, h = g & 15;
    const int n0 = b * (NUM_BLOCKS / BATCH);
    float gmax = -3.402823466e38f;
#pragma unroll
    for (int i = 0; i < 32; ++i) gmax = fmaxf(gmax, bmax[(size_t)(n0 + i) * NH + h]);
    float wgt[32];
    float gsum = 0.f;
    float sadj[32];
#pragma unroll
    for (int i = 0; i < 32; ++i) {
        float a = __expf(bmax[(size_t)(n0 + i) * NH + h] - gmax);
        wgt[i] = a;
        sadj[i] = bsum[(size_t)(n0 + i) * NH + h] * a;
        gsum += sadj[i];
    }
#pragma unroll
    for (int i = 0; i < 32; ++i) wgt[i] = wgt[i] / fmaxf(gsum, sadj[i]);
    for (int v = threadIdx.x; v < KV_LORA; v += 128) {
        float o = 0.f;
#pragma unroll
        for (int i = 0; i < 32; ++i)
            o += o_part[((size_t)(n0 + i) * NH + h) * KV_LORA + v] * wgt[i];
        out_comb[((size_t)b * NH + h) * KV_LORA + v] = o;
    }
}

// ---------------------------------------------------------------------------
// Kernel 3a: up[b, h*128+j] = sum_v out[b,h,v] * W_UV[h,v,j]  (per-head GEMM)
// ---------------------------------------------------------------------------
__global__ __launch_bounds__(256) void mla_up(
    const float* __restrict__ out_comb, const float* __restrict__ W_UV,
    float* __restrict__ up)
{
    __shared__ __align__(16) _Float16 wT[V_HEAD * WTSTR];  // 133,120 B (W_UV[h]^T)
    const int h    = blockIdx.x;
    const int tid  = threadIdx.x;
    const int wave = tid >> 5;
    const int lane = tid & 31;

    const float* wsrc = W_UV + (size_t)h * KV_LORA * V_HEAD;
    for (int e4 = tid; e4 < KV_LORA * V_HEAD / 4; e4 += 256) {
        int e = e4 * 4, v = e >> 7, j = e & 127;
        float4 f = *(const float4*)(wsrc + e);
        wT[(j + 0) * WTSTR + v] = (_Float16)f.x;
        wT[(j + 1) * WTSTR + v] = (_Float16)f.y;
        wT[(j + 2) * WTSTR + v] = (_Float16)f.z;
        wT[(j + 3) * WTSTR + v] = (_Float16)f.w;
    }
    __syncthreads();

    const int crow = lane & 15, hi = lane >> 4, kb = hi * 8;
    const int col = wave * 16 + crow;  // j
#pragma unroll
    for (int mt = 0; mt < 2; ++mt) {
        v8f acc = {};
        const float* arow =
            out_comb + ((size_t)(mt * 16 + crow) * NH + h) * KV_LORA;
#pragma unroll
        for (int kc = 0; kc < KV_LORA / 32; ++kc) {
            int off = kc * 32 + kb;
            float4 a0 = *(const float4*)(arow + off);
            float4 a1 = *(const float4*)(arow + off + 4);
            float4 a2 = *(const float4*)(arow + off + 16);
            float4 a3 = *(const float4*)(arow + off + 20);
            v16h A = frag_f32(a0, a1, a2, a3);
            v16h B = frag_lds(&wT[col * WTSTR + off]);
            acc = wmma_f16(A, B, acc);
        }
#pragma unroll
        for (int j = 0; j < 8; ++j) {
            int m = mt * 16 + hi * 8 + j;  // batch
            up[(size_t)m * D_MODEL + h * V_HEAD + col] = acc[j];
        }
    }
}

// ---------------------------------------------------------------------------
// Kernel 3b: result(32x2048) = up(32x2048) @ W_O(2048x2048)
// ---------------------------------------------------------------------------
__global__ __launch_bounds__(256) void mla_proj(
    const float* __restrict__ up, const float* __restrict__ W_O,
    float* __restrict__ result)
{
    __shared__ __align__(16) _Float16 aL[32 * USTR];  // 131,584 B
    const int tid  = threadIdx.x;
    const int wave = tid >> 5;
    const int lane = tid & 31;

    for (int e4 = tid; e4 < 32 * D_MODEL / 4; e4 += 256) {
        int e = e4 * 4, bb = e >> 11, d = e & 2047;
        float4 f = *(const float4*)(up + (size_t)bb * D_MODEL + d);
        aL[bb * USTR + d + 0] = (_Float16)f.x;
        aL[bb * USTR + d + 1] = (_Float16)f.y;
        aL[bb * USTR + d + 2] = (_Float16)f.z;
        aL[bb * USTR + d + 3] = (_Float16)f.w;
    }
    __syncthreads();

    const int crow = lane & 15, hi = lane >> 4, kb = hi * 8;
    const int col = (blockIdx.x * 8 + wave) * 16 + crow;  // output channel
    v8f acc0 = {}, acc1 = {};
#pragma unroll 4
    for (int kc = 0; kc < D_MODEL / 32; ++kc) {
        v16h B;
        const float* base = W_O + (size_t)(kc * 32 + kb) * D_MODEL + col;
#pragma unroll
        for (int i = 0; i < 8; ++i) B[i] = (_Float16)base[(size_t)i * D_MODEL];
        const float* base2 = base + (size_t)16 * D_MODEL;
#pragma unroll
        for (int i = 0; i < 8; ++i) B[8 + i] = (_Float16)base2[(size_t)i * D_MODEL];
        v16h A0 = frag_lds(&aL[crow * USTR + kc * 32 + kb]);
        acc0 = wmma_f16(A0, B, acc0);
        v16h A1 = frag_lds(&aL[(16 + crow) * USTR + kc * 32 + kb]);
        acc1 = wmma_f16(A1, B, acc1);
    }
#pragma unroll
    for (int j = 0; j < 8; ++j) {
        result[(size_t)(hi * 8 + j) * D_MODEL + col] = acc0[j];
        result[(size_t)(16 + hi * 8 + j) * D_MODEL + col] = acc1[j];
    }
}

// ---------------------------------------------------------------------------
extern "C" void kernel_launch(void* const* d_in, const int* in_sizes, int n_in,
                              void* d_out, int out_size, void* d_ws,
                              size_t ws_size, hipStream_t stream)
{
    const float* q_nope      = (const float*)d_in[0];
    const float* q_pe        = (const float*)d_in[1];
    const float* key_cache   = (const float*)d_in[2];
    const float* value_cache = (const float*)d_in[3];
    const float* W_UV        = (const float*)d_in[4];
    const float* W_O         = (const float*)d_in[5];
    const float* block_bias  = (const float*)d_in[6];
    const int*   block_list  = (const int*)d_in[7];
    const int*   block_grp   = (const int*)d_in[8];
    float* out = (float*)d_out;

    float* ws       = (float*)d_ws;
    float* o_part   = ws;                                          // 1024*16*512
    float* bmaxp    = o_part + (size_t)NUM_BLOCKS * NH * KV_LORA;  // 1024*16
    float* bsump    = bmaxp + (size_t)NUM_BLOCKS * NH;             // 1024*16
    float* out_comb = bsump + (size_t)NUM_BLOCKS * NH;             // 32*16*512
    float* upbuf    = out_comb + (size_t)BATCH * NH * KV_LORA;     // 32*2048

    mla_block_attn<<<NUM_BLOCKS, 256, 0, stream>>>(
        q_nope, q_pe, key_cache, value_cache, block_bias, block_list, block_grp,
        o_part, bmaxp, bsump);
    mla_combine<<<BATCH * NH, 128, 0, stream>>>(o_part, bmaxp, bsump, out_comb);
    mla_up<<<NH, 256, 0, stream>>>(out_comb, W_UV, upbuf);
    mla_proj<<<D_MODEL / V_HEAD, 256, 0, stream>>>(upbuf, W_O, out);
}
